// CompleteBoltzmannSolver_27977416966523
// MI455X (gfx1250) — matrix operand order repaired and checked by reading, not compiled, not run on hardware
//
#include <hip/hip_runtime.h>
#include <hip/hip_bf16.h>
#include <math.h>
#include <stdint.h>

// ---------------------------------------------------------------------------
// CDNA5 / gfx1250 implementation of the Boltzmann collision step.
// Dominant cost: 4-layer 3D CNN -> implicit GEMM with V_WMMA_F32_16X16X32_F16,
// async global->LDS staging of im2col tiles, register-hoisted B-offset tables.
// ---------------------------------------------------------------------------

typedef _Float16 half_t;
typedef __attribute__((ext_vector_type(16))) _Float16     v16h;
typedef __attribute__((ext_vector_type(8)))  float        v8f;
typedef __attribute__((ext_vector_type(8)))  unsigned int v8u;

#define NVQ    16
#define NVOX   4096          // 16^3 velocity voxels
#define NCELL  64            // 4^3 spatial cells (B=1)
#define NELEM  262144        // NVOX * NCELL
#define DVc    0.6666666667f // 10/15
#define V0c    (-5.0f)
#define DV3c   0.29629629629f
#define PI_F   3.14159265358979f
#define PI4_F  0.78539816339745f   // pi/4 : dealiased spectral d/dx on 4-pt axis

// ---------------------------------------------------------------------------
// block reduction helper (blockDim must be power of two, <=256 here)
// ---------------------------------------------------------------------------
__device__ __forceinline__ float block_reduce(float v, float* sm) {
  int tid = threadIdx.x;
  sm[tid] = v; __syncthreads();
  for (int s = blockDim.x >> 1; s > 0; s >>= 1) {
    if (tid < s) sm[tid] += sm[tid + s];
    __syncthreads();
  }
  float r = sm[0]; __syncthreads();
  return r;
}

// ---------------------------------------------------------------------------
// moments: per spatial cell, reduce over 4096 velocity voxels.
// f layout: [vox*64 + cell].  out: mom[cell*8 + {0:n,1:ux,2:uy,3:uz,4:T}]
// ---------------------------------------------------------------------------
__global__ void moments_kernel(const float* __restrict__ f, float* __restrict__ mom) {
  __shared__ float sm[256];
  const int cell = blockIdx.x;
  float s0 = 0.f, sx = 0.f, sy = 0.f, sz = 0.f, sq = 0.f;
  for (int vox = threadIdx.x; vox < NVOX; vox += blockDim.x) {
    float val = f[(size_t)vox * NCELL + cell];
    int ix = vox >> 8, iy = (vox >> 4) & 15, iz = vox & 15;
    float vx = V0c + ix * DVc, vy = V0c + iy * DVc, vz = V0c + iz * DVc;
    s0 += val; sx += vx * val; sy += vy * val; sz += vz * val;
    sq += (vx * vx + vy * vy + vz * vz) * val;
  }
  s0 = block_reduce(s0, sm);
  sx = block_reduce(sx, sm);
  sy = block_reduce(sy, sm);
  sz = block_reduce(sz, sm);
  sq = block_reduce(sq, sm);
  if (threadIdx.x == 0) {
    float dens = s0 * DV3c;
    float mx = sx * DV3c, my = sy * DV3c, mz = sz * DV3c;
    float inv = 1.f / (dens + 1e-10f);
    float ux = mx * inv, uy = my * inv, uz = mz * inv;
    float ke = (sq - 2.f * (ux * sx + uy * sy + uz * sz)
               + (ux * ux + uy * uy + uz * uz) * s0) * DV3c;
    float T = 2.f * ke / (3.f * (dens + 1e-10f));
    mom[cell * 8 + 0] = dens;
    mom[cell * 8 + 1] = ux;
    mom[cell * 8 + 2] = uy;
    mom[cell * 8 + 3] = uz;
    mom[cell * 8 + 4] = T;
  }
}

// ---------------------------------------------------------------------------
// spectral advection (4-point dealiased axis reduces to (pi/4)(f[+1]-f[-1]))
// ---------------------------------------------------------------------------
__global__ void advect_kernel(const float* __restrict__ f, const float* __restrict__ mom,
                              const float* __restrict__ dt_p, float* __restrict__ fadv) {
  int idx = blockIdx.x * blockDim.x + threadIdx.x;
  if (idx >= NELEM) return;
  int cell = idx & 63;
  int vox  = idx >> 6;
  int x = cell >> 4, y = (cell >> 2) & 3, z = cell & 3;
  float dt = dt_p[0];
  float ux = mom[cell * 8 + 1], uy = mom[cell * 8 + 2], uz = mom[cell * 8 + 3];
  size_t vb = (size_t)vox * NCELL;
  float fxp = f[vb + (((x + 1) & 3) << 4) + (y << 2) + z];
  float fxm = f[vb + (((x + 3) & 3) << 4) + (y << 2) + z];
  float fyp = f[vb + (x << 4) + (((y + 1) & 3) << 2) + z];
  float fym = f[vb + (x << 4) + (((y + 3) & 3) << 2) + z];
  float fzp = f[vb + (x << 4) + (y << 2) + ((z + 1) & 3)];
  float fzm = f[vb + (x << 4) + (y << 2) + ((z + 3) & 3)];
  float dfdx = PI4_F * (fxp - fxm);
  float dfdy = PI4_F * (fyp - fym);
  float dfdz = PI4_F * (fzp - fzm);
  fadv[idx] = f[idx] - dt * (ux * dfdx + uy * dfdy + uz * dfdz);
}

// ---------------------------------------------------------------------------
// force term: finite-difference gradients over velocity axes of f_adv
// ---------------------------------------------------------------------------
__device__ __forceinline__ float fd_axis(const float* fadv, int vox, int cell,
                                         int i, int stride) {
  size_t b = ((size_t)vox << 6) + cell;
  if (i == 0)       return (fadv[b + (size_t)stride * 64] - fadv[b]) / DVc;
  else if (i == 15) return (fadv[b] - fadv[b - (size_t)stride * 64]) / DVc;
  else              return (fadv[b + (size_t)stride * 64] - fadv[b - (size_t)stride * 64])
                            / (2.f * DVc);
}

__global__ void force_kernel(const float* __restrict__ fadv, const float* __restrict__ forces,
                             const float* __restrict__ dt_p, float* __restrict__ fforc) {
  int idx = blockIdx.x * blockDim.x + threadIdx.x;
  if (idx >= NELEM) return;
  int cell = idx & 63;
  int vox  = idx >> 6;
  int ix = vox >> 8, iy = (vox >> 4) & 15, iz = vox & 15;
  float dt = dt_p[0];
  float fx = forces[0 * 64 + cell];
  float fy = forces[1 * 64 + cell];
  float fz = forces[2 * 64 + cell];
  float gx = fd_axis(fadv, vox, cell, ix, 256);
  float gy = fd_axis(fadv, vox, cell, iy, 16);
  float gz = fd_axis(fadv, vox, cell, iz, 1);
  fforc[idx] = fadv[idx] - dt * (fx * gx + fy * gy + fz * gz);
}

// ---------------------------------------------------------------------------
// weight packing: w[Cout][Cin][125] f32 -> wp[Mpad][Cin*128] f16
// (per-cin K padded 125 -> 128 so every WMMA K-step stays within one cin)
// ---------------------------------------------------------------------------
__global__ void pack_weights_kernel(const float* __restrict__ w, half_t* __restrict__ wp,
                                    int Cout, int Cin, int Mpad) {
  int total = Mpad * (Cin << 7);
  for (int idx = blockIdx.x * blockDim.x + threadIdx.x; idx < total;
       idx += gridDim.x * blockDim.x) {
    int m   = idx >> 7;          // combined (mrow, cin): idx = ((mrow*Cin)+cin)*128 + kr
    int kr  = idx & 127;
    int cin = m % Cin;
    int mr  = m / Cin;
    half_t v = (half_t)0.f;
    if (mr < Cout && kr < 125)
      v = (half_t)w[((size_t)mr * Cin + cin) * 125 + kr];
    wp[idx] = v;
  }
}

// f_forced [vox*64+cell] -> activation f16 [sample*4096 + vox]
__global__ void pack_input_kernel(const float* __restrict__ ff, half_t* __restrict__ act) {
  int idx = blockIdx.x * blockDim.x + threadIdx.x;
  if (idx >= NELEM) return;
  int s = idx >> 12, vox = idx & 4095;
  act[idx] = (half_t)ff[(size_t)vox * NCELL + s];
}

// ---------------------------------------------------------------------------
// Implicit-GEMM 3D conv (5x5x5, pad 2) via WMMA f16 -> f32.
// Block = 128 threads = 4 wave32. Block tile: 16 couts x 64 positions
// (one contiguous vox chunk: fixed d, h0..h0+3 (one h per wave), w 0..15).
// Steady-state loop per cin:
//   <=4 predicated global_load_async_to_lds_b32 (precomputed descriptors)
//   s_wait_asynccnt 0 + barrier
//   4 x { 8 dword A loads, 16 ds_load_u16 B reads (hoisted offsets), WMMA }
// ---------------------------------------------------------------------------
#define TILE_H 8
#define TILE_W 20
#define TILE_VALID 800          // 5*8*20
#define TILE_ZBASE 800          // zero slots [800,1104)
#define TILE_SZ   1104

__global__ void __launch_bounds__(128)
conv_wmma_kernel(const half_t* __restrict__ in, const half_t* __restrict__ wp,
                 const float* __restrict__ bias, half_t* __restrict__ out,
                 int Cin, int Cout) {
  __shared__ half_t tile[TILE_SZ];
  __shared__ unsigned short tbl[128];

  const int tid  = threadIdx.x;
  const int lane = tid & 31;
  const int wave = tid >> 5;           // = local h offset (0..3)
  const int hi   = lane >> 4;          // half-wave select
  const int ml   = lane & 15;          // A row within tile
  const int n    = lane & 15;          // B/C column within tile (= w)
  const int m0   = blockIdx.x << 4;

  const int vox_base = (blockIdx.y << 6) & 4095;
  const int s  = blockIdx.y >> 6;      // sample
  const int d  = vox_base >> 8;        // fixed depth for block
  const int h0 = (vox_base >> 4) & 15; // 4-aligned h base
  const int vox = vox_base + (wave << 4) + n;

  const int Kpad = Cin << 7;
  const half_t* __restrict__ arow = wp + (size_t)(m0 + ml) * Kpad;
  const int hlw = wave * TILE_W + n;   // lane's (h_local, w) offset in tile

  // ---- one-time LDS init: zero tile (incl. zero slots) + kr offset table ----
  for (int i = tid; i < TILE_SZ; i += 128) tile[i] = (half_t)0.f;
  {
    int kr = tid;                      // blockDim == 128
    unsigned short off = TILE_ZBASE;
    if (kr < 125) {
      int kd = kr / 25;
      int r  = kr - kd * 25;
      int kh = r / 5;
      int kw = r - kh * 5;
      off = (unsigned short)(kd * (TILE_H * TILE_W) + kh * TILE_W + kw);
    }
    tbl[kr] = off;
  }
  __syncthreads();

  // ---- hoist B-fragment tile offsets into registers (cin-invariant) ----
  int toff[64];
#pragma unroll
  for (int kk = 0; kk < 4; ++kk) {
    const unsigned short* tp = &tbl[(kk << 5) + (hi << 4)];
#pragma unroll
    for (int j = 0; j < 16; ++j)
      toff[kk * 16 + j] = (int)tp[j] + hlw;
  }

  // ---- precompute async staging descriptors (cin-invariant geometry) ----
  uint32_t st_lds[4];
  int      st_go[4];
  bool     st_in[4];
#pragma unroll
  for (int it = 0; it < 4; ++it) {
    int slot = tid + it * 128;
    bool valid = slot < (TILE_VALID / 2);
    int dz = 0, hh = 0, wpair = 0;
    if (valid) {
      dz    = slot / 80;               // 80 dword slots per depth plane
      int r = slot - dz * 80;
      hh    = r / 10;                  // 10 dword pairs per row
      wpair = r - hh * 10;
    }
    int zd = d + dz - 2;
    int zh = h0 + hh - 2;
    int zw = (wpair << 1) - 2;
    st_in[it]  = valid && ((unsigned)zd < 16u) && ((unsigned)zh < 16u) &&
                 (wpair >= 1) && (wpair <= 8);
    st_lds[it] = (uint32_t)(uintptr_t)(&tile[slot << 1]);
    st_go[it]  = (zd << 8) + (zh << 4) + zw;
  }

  v8f acc = {};

  for (int cin = 0; cin < Cin; ++cin) {
    // ---- async stage of this cin's halo region ----
    const half_t* __restrict__ src = in + ((size_t)(s * Cin + cin) << 12);
#pragma unroll
    for (int it = 0; it < 4; ++it) {
      if (st_in[it]) {
        const half_t* gaddr = src + st_go[it];
        asm volatile("global_load_async_to_lds_b32 %0, %1, off"
                     :: "v"(st_lds[it]), "v"(gaddr) : "memory");
      }
    }
    asm volatile("s_wait_asynccnt 0x0" ::: "memory");
    __syncthreads();

    const half_t* __restrict__ wrow = arow + (cin << 7);
    __builtin_prefetch(wrow + 128, 0, 0);       // next cin's weight slab

#pragma unroll
    for (int kk = 0; kk < 4; ++kk) {
      // --- A fragment: 8 aligned dwords, contiguous k = 16*hi + s (matches B) ---
      const uint32_t* w32 = (const uint32_t*)(wrow + (kk << 5));
      v8u au;
#pragma unroll
      for (int i = 0; i < 8; ++i)
        au[i] = w32[(hi << 3) + i];
      v16h a = __builtin_bit_cast(v16h, au);

      // --- B fragment: independent LDS reads at hoisted offsets ---
      v16h b;
#pragma unroll
      for (int j = 0; j < 16; ++j)
        b[j] = tile[toff[kk * 16 + j]];

      acc = __builtin_amdgcn_wmma_f32_16x16x32_f16(
          false, a, false, b, (short)0, acc, false, false);
    }
    __syncthreads();                  // waves done reading before next stage
  }

#pragma unroll
  for (int r = 0; r < 8; ++r) {
    int m = m0 + r + (hi << 3);
    if (m < Cout)
      out[((size_t)(s * Cout + m) << 12) + vox] = (half_t)(acc[r] + bias[m]);
  }
}

// ---------------------------------------------------------------------------
// GroupNorm + SiLU, in place on f16 activations. block = (sample, group)
// ---------------------------------------------------------------------------
__global__ void gn_silu_kernel(half_t* __restrict__ buf, const float* __restrict__ g,
                               const float* __restrict__ be, int C, int groups) {
  __shared__ float sm[256];
  __shared__ float mu_s, rstd_s;
  const int s   = blockIdx.x;
  const int grp = blockIdx.y;
  const int Cg  = C / groups;
  const size_t base  = ((size_t)s * C + (size_t)grp * Cg) << 12;
  const int    nelem = Cg << 12;
  float sum = 0.f, ssq = 0.f;
  for (int i = threadIdx.x; i < nelem; i += blockDim.x) {
    float v = (float)buf[base + i];
    sum += v; ssq += v * v;
  }
  sum = block_reduce(sum, sm);
  ssq = block_reduce(ssq, sm);
  if (threadIdx.x == 0) {
    float mu  = sum / (float)nelem;
    float var = ssq / (float)nelem - mu * mu;
    mu_s = mu; rstd_s = rsqrtf(var + 1e-5f);
  }
  __syncthreads();
  for (int i = threadIdx.x; i < nelem; i += blockDim.x) {
    int c = grp * Cg + (i >> 12);
    float v = ((float)buf[base + i] - mu_s) * rstd_s * g[c] + be[c];
    float y = v / (1.f + __expf(-v));       // SiLU
    buf[base + i] = (half_t)y;
  }
}

// ---------------------------------------------------------------------------
// conservation pass 1: c = bgk + 0.1*nc ; accumulate M0, M1x/y/z per cell
// ---------------------------------------------------------------------------
__global__ void final1_kernel(const float* __restrict__ ff, const half_t* __restrict__ nc,
                              const float* __restrict__ mom2, const float* __restrict__ tau_p,
                              float* __restrict__ cbuf, float* __restrict__ sums) {
  __shared__ float sm[256];
  const int cell = blockIdx.x;
  const float dens = mom2[cell * 8 + 0];
  const float ux = mom2[cell * 8 + 1], uy = mom2[cell * 8 + 2], uz = mom2[cell * 8 + 3];
  const float T  = mom2[cell * 8 + 4];
  const float inv_tau = 1.f / tau_p[0];
  const float pref = dens * powf(2.f * PI_F * T + 1e-10f, -1.5f);
  const float inv2T = 1.f / (2.f * T + 1e-10f);
  float m0 = 0.f, mx = 0.f, my = 0.f, mz = 0.f;
  for (int vox = threadIdx.x; vox < NVOX; vox += blockDim.x) {
    int ix = vox >> 8, iy = (vox >> 4) & 15, iz = vox & 15;
    float vx = V0c + ix * DVc, vy = V0c + iy * DVc, vz = V0c + iz * DVc;
    float rx = vx - ux, ry = vy - uy, rz = vz - uz;
    float ex = -(rx * rx + ry * ry + rz * rz) * inv2T;
    ex = fminf(fmaxf(ex, -50.f), 50.f);
    float feq = pref * __expf(ex);
    float fv  = ff[(size_t)vox * NCELL + cell];
    float c   = -(fv - feq) * inv_tau + 0.1f * (float)nc[((size_t)cell << 12) + vox];
    cbuf[((size_t)cell << 12) + vox] = c;
    m0 += c; mx += c * vx; my += c * vy; mz += c * vz;
  }
  m0 = block_reduce(m0, sm);
  mx = block_reduce(mx, sm);
  my = block_reduce(my, sm);
  mz = block_reduce(mz, sm);
  if (threadIdx.x == 0) {
    sums[cell * 8 + 0] = m0 * DV3c;
    sums[cell * 8 + 1] = mx * DV3c;
    sums[cell * 8 + 2] = my * DV3c;
    sums[cell * 8 + 3] = mz * DV3c;
  }
}

// pass 2: subtract mass & momentum projections, accumulate M2u
__global__ void final2_kernel(float* __restrict__ cbuf, const float* __restrict__ sums,
                              float* __restrict__ m2u) {
  __shared__ float sm[256];
  const int cell = blockIdx.x;
  float S2 = 0.f;
  for (int i = 0; i < 16; ++i) { float v = V0c + i * DVc; S2 += v * v; }
  const float Sv2 = 256.f * S2 * DV3c;              // sum(vx^2)*DV3 over grid
  const float c0 = sums[cell * 8 + 0] / ((float)NVOX * DV3c);
  const float cx = sums[cell * 8 + 1] / (Sv2 + 1e-10f);
  const float cy = sums[cell * 8 + 2] / (Sv2 + 1e-10f);
  const float cz = sums[cell * 8 + 3] / (Sv2 + 1e-10f);
  float m2 = 0.f;
  for (int vox = threadIdx.x; vox < NVOX; vox += blockDim.x) {
    int ix = vox >> 8, iy = (vox >> 4) & 15, iz = vox & 15;
    float vx = V0c + ix * DVc, vy = V0c + iy * DVc, vz = V0c + iz * DVc;
    float c = cbuf[((size_t)cell << 12) + vox] - c0 - cx * vx - cy * vy - cz * vz;
    cbuf[((size_t)cell << 12) + vox] = c;
    m2 += c * (vx * vx + vy * vy + vz * vz);
  }
  m2 = block_reduce(m2, sm);
  if (threadIdx.x == 0) m2u[cell] = m2 * DV3c;
}

// pass 3: energy projection, relu, write output in reference layout
__global__ void final3_kernel(const float* __restrict__ ff, const float* __restrict__ cbuf,
                              const float* __restrict__ m2u, const float* __restrict__ dt_p,
                              float* __restrict__ out) {
  int idx = blockIdx.x * blockDim.x + threadIdx.x;
  if (idx >= NELEM) return;
  int cell = idx & 63, vox = idx >> 6;
  int ix = vox >> 8, iy = (vox >> 4) & 15, iz = vox & 15;
  float vx = V0c + ix * DVc, vy = V0c + iy * DVc, vz = V0c + iz * DVc;
  float vsq = vx * vx + vy * vy + vz * vz;
  float S2 = 0.f, S4 = 0.f;
  for (int i = 0; i < 16; ++i) {
    float v = V0c + i * DVc; float v2 = v * v;
    S2 += v2; S4 += v2 * v2;
  }
  float Svq2 = (3.f * 256.f * S4 + 6.f * 16.f * S2 * S2) * DV3c;  // sum(v_sq^2)*DV3
  float c = cbuf[((size_t)cell << 12) + vox] - m2u[cell] * vsq / (Svq2 + 1e-10f);
  float r = ff[idx] + dt_p[0] * c;
  out[idx] = fmaxf(r, 0.f);
}

// ---------------------------------------------------------------------------
// host launcher
// ---------------------------------------------------------------------------
extern "C" void kernel_launch(void* const* d_in, const int* in_sizes, int n_in,
                              void* d_out, int out_size, void* d_ws, size_t ws_size,
                              hipStream_t stream) {
  (void)in_sizes; (void)n_in; (void)out_size; (void)ws_size;
  const float* f      = (const float*)d_in[0];
  const float* forces = (const float*)d_in[1];
  const float* dt     = (const float*)d_in[2];
  const float* w1 = (const float*)d_in[3];
  const float* b1 = (const float*)d_in[4];
  const float* g1 = (const float*)d_in[5];
  const float* be1 = (const float*)d_in[6];
  const float* w2 = (const float*)d_in[7];
  const float* b2 = (const float*)d_in[8];
  const float* g2 = (const float*)d_in[9];
  const float* be2 = (const float*)d_in[10];
  const float* w3 = (const float*)d_in[11];
  const float* b3 = (const float*)d_in[12];
  const float* g3 = (const float*)d_in[13];
  const float* be3 = (const float*)d_in[14];
  const float* w4 = (const float*)d_in[15];
  const float* b4 = (const float*)d_in[16];
  const float* tau = (const float*)d_in[17];
  float* out = (float*)d_out;

  char* ws = (char*)d_ws;
  float*  f_adv  = (float*)(ws + (size_t)0);
  float*  f_forc = (float*)(ws + ((size_t)1 << 20));
  float*  cbuf   = (float*)(ws + ((size_t)2 << 20));
  float*  mom1   = (float*)(ws + ((size_t)3 << 20));
  float*  mom2   = (float*)(ws + ((size_t)3 << 20) + 4096);
  float*  sums   = (float*)(ws + ((size_t)3 << 20) + 8192);
  float*  m2u    = (float*)(ws + ((size_t)3 << 20) + 12288);
  half_t* wp1    = (half_t*)(ws + ((size_t)4 << 20));   // 64   x 128   f16
  half_t* wp2    = (half_t*)(ws + ((size_t)5 << 20));   // 128  x 8192  f16 = 2MB
  half_t* wp3    = (half_t*)(ws + ((size_t)7 << 20));   // 64   x 16384 f16 = 2MB
  half_t* wp4    = (half_t*)(ws + ((size_t)9 << 20));   // 16   x 8192  f16
  half_t* bufA   = (half_t*)(ws + ((size_t)10 << 20));  // 64 MB
  half_t* bufB   = (half_t*)(ws + ((size_t)74 << 20));  // 64 MB

  // ---- weight packing: per-cin K 125 -> 128, M padded to WMMA tile ----
  pack_weights_kernel<<<32, 256, 0, stream>>>(w1, wp1, 64, 1, 64);
  pack_weights_kernel<<<4096, 256, 0, stream>>>(w2, wp2, 128, 64, 128);
  pack_weights_kernel<<<4096, 256, 0, stream>>>(w3, wp3, 64, 128, 64);
  pack_weights_kernel<<<512, 256, 0, stream>>>(w4, wp4, 1, 64, 16);

  // ---- physics: moments -> advection -> force term ----
  moments_kernel<<<NCELL, 256, 0, stream>>>(f, mom1);
  advect_kernel<<<NELEM / 256, 256, 0, stream>>>(f, mom1, dt, f_adv);
  force_kernel<<<NELEM / 256, 256, 0, stream>>>(f_adv, forces, dt, f_forc);
  moments_kernel<<<NCELL, 256, 0, stream>>>(f_forc, mom2);

  // ---- collision network: 4 WMMA implicit-GEMM convs + GN/SiLU ----
  pack_input_kernel<<<NELEM / 256, 256, 0, stream>>>(f_forc, bufA);
  conv_wmma_kernel<<<dim3(4, 4096), 128, 0, stream>>>(bufA, wp1, b1, bufB, 1, 64);
  gn_silu_kernel<<<dim3(64, 8), 256, 0, stream>>>(bufB, g1, be1, 64, 8);
  conv_wmma_kernel<<<dim3(8, 4096), 128, 0, stream>>>(bufB, wp2, b2, bufA, 64, 128);
  gn_silu_kernel<<<dim3(64, 16), 256, 0, stream>>>(bufA, g2, be2, 128, 16);
  conv_wmma_kernel<<<dim3(4, 4096), 128, 0, stream>>>(bufA, wp3, b3, bufB, 128, 64);
  gn_silu_kernel<<<dim3(64, 8), 256, 0, stream>>>(bufB, g3, be3, 64, 8);
  conv_wmma_kernel<<<dim3(1, 4096), 128, 0, stream>>>(bufB, wp4, b4, bufA, 64, 1);

  // ---- BGK + neural collision, conservation projection, relu ----
  final1_kernel<<<NCELL, 256, 0, stream>>>(f_forc, bufA, mom2, tau, cbuf, sums);
  final2_kernel<<<NCELL, 256, 0, stream>>>(cbuf, sums, m2u);
  final3_kernel<<<NELEM / 256, 256, 0, stream>>>(f_forc, cbuf, m2u, dt, out);
}
